// TemporalSpectrumBlock_23622320128094
// MI455X (gfx1250) — matrix operand order repaired and checked by reading, not compiled, not run on hardware
//
#include <hip/hip_runtime.h>
#include <hip/hip_bf16.h>
#include <math.h>

// ---------------------------------------------------------------------------
// VMamba-style block on gfx1250 (wave32, WMMA, async-to-LDS).
// Sizes: B=2, H=W=64 -> L=4096, C=96, Din=192, Nstate=16, R=6, K=4 dirs.
// GEMMs use v_wmma_f32_16x16x32_f16 (f16 operands, f32 accum).
// Selective scan uses a 3-phase chunked scan (64 chunks of 64) to trade
// 2x v_exp work for 64x parallelism on the sequential critical path.
// Scan B/C staging uses GLOBAL_LOAD_ASYNC_TO_LDS_B128 + s_wait_asynccnt.
// ---------------------------------------------------------------------------

typedef _Float16 f16;
typedef _Float16 v16h __attribute__((ext_vector_type(16)));
typedef _Float16 v8h  __attribute__((ext_vector_type(8)));
typedef float    v8f  __attribute__((ext_vector_type(8)));

#define LSEQ 4096
#define NC_  64
#define CS_  64

__device__ __forceinline__ v8f wmma16(v16h a, v16h b, v8f c) {
  // (neg_a, A, neg_b, B, c_mod, C, reuse_a, reuse_b)
  return __builtin_amdgcn_wmma_f32_16x16x32_f16(false, a, false, b, (short)0, c, false, false);
}

// A-operand loader: contiguous k order (B uses the same) -> per lane two contiguous 8-half
// groups at p and p+8 (k = 16*hi + j).
__device__ __forceinline__ v16h ldA(const f16* p) {
  v8h lo = *(const v8h*)p;
  v8h hi = *(const v8h*)(p + 8);
  v16h r;
#pragma unroll
  for (int i = 0; i < 8; ++i) { r[i] = lo[i]; r[i + 8] = hi[i]; }
  return r;
}

// Async copy 16B global -> 16B LDS (ASYNCcnt-tracked, no VGPR round trip).
// GVS addressing: mem = SADDR(64b sgpr) + VADDR(32b vgpr); VDST = LDS byte addr.
__device__ __forceinline__ void async_ld_b128(unsigned lds_off, const float* sbase,
                                              unsigned goff) {
  asm volatile("global_load_async_to_lds_b128 %0, %1, %2"
               :: "v"(lds_off), "v"(goff), "s"(sbase) : "memory");
}
__device__ __forceinline__ void wait_async0() {
  asm volatile("s_wait_asynccnt 0x0" ::: "memory");
}

// direction k, sequence position l -> memory index (row-major h*64+w).
__device__ __forceinline__ int lmapf(int k, int l) {
  int ls = (k >= 2) ? (LSEQ - 1 - l) : l;           // dirs 2,3 are flips
  if (k & 1) return (ls & 63) * 64 + (ls >> 6);     // dirs 1,3 are transposed
  return ls;
}

// --- K0: pack weights (f16 copies, zero-padded tiles, A = -exp(A_log)) -----
__global__ void k_pack(const float* inw, const float* outw, const float* xpw,
                       const float* dtw, const float* alog,
                       f16* win, f16* wout, f16* wxp, f16* wdt, float* Aneg) {
  int id = blockIdx.x * 256 + threadIdx.x;
  if (id < 36864) { win[id] = (f16)inw[id]; return; }
  id -= 36864;
  if (id < 18432) { wout[id] = (f16)outw[id]; return; }
  id -= 18432;
  if (id < 36864) {                       // x_proj padded (4,48,192)
    int k = id / 9216, rem = id % 9216, c = rem / 192, d = rem % 192;
    wxp[id] = (c < 38) ? (f16)xpw[(k * 38 + c) * 192 + d] : (f16)0.f;
    return;
  }
  id -= 36864;
  if (id < 24576) {                       // dt_w padded (4,192,32)
    int k = id / 6144, rem = id % 6144, d = rem / 32, r = rem % 32;
    wdt[id] = (r < 6) ? (f16)dtw[(k * 192 + d) * 6 + r] : (f16)0.f;
    return;
  }
  id -= 24576;
  if (id < 12288) Aneg[id] = -__expf(alog[id]);
}

// --- K1: diff = |x-y| (store f32) + LayerNorm -> f16 activations -----------
__global__ void k_lndiff(const float* x, const float* y, const float* lnw,
                         const float* lnb, float* diff, f16* aln) {
  int wv = (blockIdx.x * blockDim.x + threadIdx.x) >> 5;   // row in [0,8192)
  int lane = threadIdx.x & 31;
  const float* xr = x + wv * 96;
  const float* yr = y + wv * 96;
  float v[3], s = 0.f, ss = 0.f;
#pragma unroll
  for (int i = 0; i < 3; ++i) {
    int c = lane + 32 * i;
    float d = fabsf(xr[c] - yr[c]);
    v[i] = d; s += d; ss += d * d;
    diff[wv * 96 + c] = d;
  }
#pragma unroll
  for (int o = 16; o > 0; o >>= 1) { s += __shfl_xor(s, o, 32); ss += __shfl_xor(ss, o, 32); }
  float mu = s / 96.f;
  float rs = rsqrtf(ss / 96.f - mu * mu + 1e-5f);
#pragma unroll
  for (int i = 0; i < 3; ++i) {
    int c = lane + 32 * i;
    aln[wv * 96 + c] = (f16)((v[i] - mu) * rs * lnw[c] + lnb[c]);
  }
}

// --- K2: in_proj GEMM (M=8192,K=96,N=384) via WMMA -------------------------
__global__ void k_inproj(const f16* aln, const f16* win, float* xm, float* z) {
  int wv = (blockIdx.x * blockDim.x + threadIdx.x) >> 5;   // 12288 tiles
  int lane = threadIdx.x & 31;
  int nt = wv % 24, mt = wv / 24;
  int lo = lane & 15, hi = lane >> 4;
  const f16* ap = aln + (mt * 16 + lo) * 96 + hi * 16;
  const f16* bp = win + (nt * 16 + lo) * 96 + hi * 16;     // B rows = out chan
  v8f acc = {};
#pragma unroll
  for (int kc = 0; kc < 3; ++kc)
    acc = wmma16(ldA(ap + kc * 32), *(const v16h*)(bp + kc * 32), acc);
  if (nt < 12) {                                           // xm half (uniform)
    int n = nt * 16 + lo;
#pragma unroll
    for (int r = 0; r < 8; ++r) {
      int m = mt * 16 + r + 8 * hi;
      int b = m >> 12, l = m & 4095;
      xm[(b * 192 + n) * LSEQ + l] = acc[r];               // NCHW for conv
    }
  } else {                                                 // z half
    int n = nt * 16 + lo - 192;
#pragma unroll
    for (int r = 0; r < 8; ++r) {
      int m = mt * 16 + r + 8 * hi;
      z[m * 192 + n] = acc[r];                             // (B,L,192)
    }
  }
}

// --- K3: depthwise 3x3 conv + bias + SiLU; emit f32 (B,L,D) + f16 dirs 0/1 -
__global__ void k_conv(const float* xm, const float* cw, const float* cb,
                       float* xmem, f16* xs0, f16* xs1) {
  __shared__ float sh[18 * 18];
  int bid = blockIdx.x;
  int tile = bid & 15, pd = bid >> 4;         // pd = b*192 + d
  int d = pd % 192, b = pd / 192;
  int ty0 = (tile >> 2) * 16, tx0 = (tile & 3) * 16;
  const float* plane = xm + pd * LSEQ;
  for (int i = threadIdx.x; i < 324; i += 256) {
    int r = i / 18, c = i % 18;
    int h = ty0 + r - 1, w = tx0 + c - 1;
    sh[i] = (h >= 0 && h < 64 && w >= 0 && w < 64) ? plane[h * 64 + w] : 0.f;
  }
  __syncthreads();
  int tx = threadIdx.x & 15, ty = threadIdx.x >> 4;
  float acc = cb[d];
#pragma unroll
  for (int kh = 0; kh < 3; ++kh)
#pragma unroll
    for (int kw = 0; kw < 3; ++kw)
      acc += sh[(ty + kh) * 18 + (tx + kw)] * cw[d * 9 + kh * 3 + kw];
  float v = acc / (1.f + __expf(-acc));       // SiLU
  int h = ty0 + ty, w = tx0 + tx;
  int lm = h * 64 + w, ls = w * 64 + h;
  xmem[(b * LSEQ + lm) * 192 + d] = v;        // scan reads (coalesced in d)
  xs0[(b * LSEQ + lm) * 192 + d] = (f16)v;    // dir0 GEMM operand
  xs1[(b * LSEQ + ls) * 192 + d] = (f16)v;    // dir1 (transposed) operand
}

// --- K4: x_dbl = x_proj_w @ xs  (M=48 padded c, K=192, N=L) via WMMA -------
__global__ void k_xdbl(const f16* xs0, const f16* xs1, const f16* wxp, float* xdbl) {
  int wv = (blockIdx.x * blockDim.x + threadIdx.x) >> 5;   // 6144 tiles
  int lane = threadIdx.x & 31;
  int ct = wv % 3, lt = (wv / 3) & 255, k = (wv / 768) & 3, b = wv / 3072;
  int lo = lane & 15, hi = lane >> 4;
  int lseq = lt * 16 + lo;
  int lsrc = (k >= 2) ? (LSEQ - 1 - lseq) : lseq;
  const f16* src = (k & 1) ? xs1 : xs0;
  const f16* bp = src + (size_t)(b * LSEQ + lsrc) * 192 + hi * 16;
  const f16* ap = wxp + (k * 48 + ct * 16 + lo) * 192 + hi * 16;
  v8f acc = {};
#pragma unroll
  for (int kc = 0; kc < 6; ++kc)
    acc = wmma16(ldA(ap + kc * 32), *(const v16h*)(bp + kc * 32), acc);
  int bk = b * 4 + k;
#pragma unroll
  for (int r = 0; r < 8; ++r) {
    int c = ct * 16 + r + 8 * hi;
    if (c < 38) xdbl[(size_t)(bk * 38 + c) * LSEQ + lseq] = acc[r];
  }
}

// --- K5: delta = softplus(dt_w @ dts + dt_b)  (M=192,K=6 pad 32,N=L) -------
__global__ void k_delta(const float* xdbl, const f16* wdt, const float* dtb,
                        float* delta) {
  int wv = (blockIdx.x * blockDim.x + threadIdx.x) >> 5;   // 24576 tiles
  int lane = threadIdx.x & 31;
  int dt_ = wv % 12, lt = (wv / 12) & 255, k = (wv / 3072) & 3, b = wv / 12288;
  int lo = lane & 15, hi = lane >> 4;
  int lseq = lt * 16 + lo;
  int bk = b * 4 + k;
  const f16* ap = wdt + k * 6144 + (dt_ * 16 + lo) * 32 + hi * 16;
  v16h bb;
#pragma unroll
  for (int j = 0; j < 16; ++j) {
    int r = hi * 16 + j;                      // k-dim index of B operand
    bb[j] = (r < 6) ? (f16)xdbl[(size_t)(bk * 38 + r) * LSEQ + lseq] : (f16)0.f;
  }
  v8f acc = {};
  acc = wmma16(ldA(ap), bb, acc);
#pragma unroll
  for (int r = 0; r < 8; ++r) {
    int d = dt_ * 16 + r + 8 * hi;
    float xv = acc[r] + dtb[k * 192 + d];
    float sp = (xv > 20.f) ? xv : log1pf(__expf(xv));
    delta[((size_t)bk * LSEQ + lseq) * 192 + d] = sp;      // (B,K,L,D)
  }
}

// --- K6: scan phase 1: per-chunk S = sum(delta*A), local state (h0 = 0) ----
// B rows staged via async-to-LDS: LDS layout [n][l] (row = 64 contiguous f32).
__global__ void k_scan1(const float* xdbl, const float* delta, const float* xmem,
                        const float* Aneg, float* Sb, float* Hl) {
  __shared__ float Bsh[16 * CS_];
  int chunk = blockIdx.x & 63, bk = blockIdx.x >> 6;
  int b = bk >> 2, k = bk & 3;
  int d = threadIdx.x;                        // 192 threads
  int c0 = chunk * CS_;
  const float* gb = xdbl + ((size_t)bk * 38 + 6) * LSEQ + c0;
  unsigned ldsb = (unsigned)(uintptr_t)(&Bsh[0]);
  for (int i = d; i < 256; i += 192)          // 16 rows x 16 chunks of 16B
    async_ld_b128(ldsb + (unsigned)i * 16, gb,
                  (unsigned)((i >> 4) * (LSEQ * 4) + (i & 15) * 16));
  float A[16], h[16], S[16];
#pragma unroll
  for (int n = 0; n < 16; ++n) {
    A[n] = Aneg[(k * 192 + d) * 16 + n];
    h[n] = 0.f; S[n] = 0.f;
  }
  wait_async0();
  __syncthreads();
  for (int il = 0; il < CS_; ++il) {
    int l = c0 + il;
    float dlt = delta[((size_t)bk * LSEQ + l) * 192 + d];
    float u = xmem[((size_t)b * LSEQ + lmapf(k, l)) * 192 + d];
    float du = dlt * u;
#pragma unroll
    for (int n = 0; n < 16; ++n) {
      float s = dlt * A[n];
      S[n] += s;
      h[n] = __expf(s) * h[n] + du * Bsh[n * CS_ + il];
    }
  }
  int base = (bk * NC_ + chunk) * 3072 + d * 16;
#pragma unroll
  for (int n = 0; n < 16; ++n) { Sb[base + n] = S[n]; Hl[base + n] = h[n]; }
}

// --- K7: scan phase 2: inter-chunk scan (64 steps, 24576 parallel states) --
__global__ void k_scan2(const float* Sb, const float* Hl, float* Hin) {
  int t = blockIdx.x * 256 + threadIdx.x;
  int bk = t / 3072, dn = t % 3072;
  float carry = 0.f;
  for (int c = 0; c < NC_; ++c) {
    int idx = (bk * NC_ + c) * 3072 + dn;
    Hin[idx] = carry;                         // state entering chunk c
    carry = __expf(Sb[idx]) * carry + Hl[idx];
  }
}

// --- K8: scan phase 3: replay chunk from Hin, emit y (+ D*u) ---------------
// B rows (c=6..21) and C rows (c=22..37) are contiguous in xdbl: stage all 32
// rows with one async-to-LDS pass. BCsh row n <-> global row 6+n.
__global__ void k_scan3(const float* xdbl, const float* delta, const float* xmem,
                        const float* Aneg, const float* Hin, const float* Ds,
                        float* ysm) {
  __shared__ float BCsh[32 * CS_];
  int chunk = blockIdx.x & 63, bk = blockIdx.x >> 6;
  int b = bk >> 2, k = bk & 3;
  int d = threadIdx.x;
  int c0 = chunk * CS_;
  const float* gb = xdbl + ((size_t)bk * 38 + 6) * LSEQ + c0;
  unsigned ldsb = (unsigned)(uintptr_t)(&BCsh[0]);
  for (int i = d; i < 512; i += 192)          // 32 rows x 16 chunks of 16B
    async_ld_b128(ldsb + (unsigned)i * 16, gb,
                  (unsigned)((i >> 4) * (LSEQ * 4) + (i & 15) * 16));
  float A[16], h[16];
  int base = (bk * NC_ + chunk) * 3072 + d * 16;
#pragma unroll
  for (int n = 0; n < 16; ++n) {
    A[n] = Aneg[(k * 192 + d) * 16 + n];
    h[n] = Hin[base + n];
  }
  float Dv = Ds[k * 192 + d];
  wait_async0();
  __syncthreads();
  for (int il = 0; il < CS_; ++il) {
    int l = c0 + il;
    float dlt = delta[((size_t)bk * LSEQ + l) * 192 + d];
    int lm = lmapf(k, l);
    float u = xmem[((size_t)b * LSEQ + lm) * 192 + d];
    float du = dlt * u;
    float y = 0.f;
#pragma unroll
    for (int n = 0; n < 16; ++n) {
      h[n] = __expf(dlt * A[n]) * h[n] + du * BCsh[n * CS_ + il];
      y += h[n] * BCsh[(16 + n) * CS_ + il];
    }
    ysm[((size_t)bk * LSEQ + lm) * 192 + d] = y + Dv * u;  // memory-space idx
  }
}

// --- K9: sum 4 directions + out LayerNorm + SiLU(z) gate -> f16 ------------
__global__ void k_combine(const float* ysm, const float* z, const float* onw,
                          const float* onb, f16* gbuf) {
  int wv = (blockIdx.x * blockDim.x + threadIdx.x) >> 5;   // (b,lmem) row
  int lane = threadIdx.x & 31;
  int b = wv >> 12, lm = wv & 4095;
  float v[6], s = 0.f, ss = 0.f;
#pragma unroll
  for (int i = 0; i < 6; ++i) {
    int d = lane + 32 * i;
    float a = 0.f;
#pragma unroll
    for (int k = 0; k < 4; ++k)
      a += ysm[((size_t)(b * 4 + k) * LSEQ + lm) * 192 + d];
    v[i] = a; s += a; ss += a * a;
  }
#pragma unroll
  for (int o = 16; o > 0; o >>= 1) { s += __shfl_xor(s, o, 32); ss += __shfl_xor(ss, o, 32); }
  float mu = s / 192.f;
  float rs = rsqrtf(ss / 192.f - mu * mu + 1e-5f);
#pragma unroll
  for (int i = 0; i < 6; ++i) {
    int d = lane + 32 * i;
    float yn = (v[i] - mu) * rs * onw[d] + onb[d];
    float zz = z[(size_t)wv * 192 + d];
    gbuf[(size_t)wv * 192 + d] = (f16)(yn * (zz / (1.f + __expf(-zz))));
  }
}

// --- K10: out_proj GEMM (M=8192,K=192,N=96) + residual ---------------------
__global__ void k_outproj(const f16* gbuf, const f16* wout, const float* diff,
                          float* out) {
  int wv = (blockIdx.x * blockDim.x + threadIdx.x) >> 5;   // 3072 tiles
  int lane = threadIdx.x & 31;
  int nt = wv % 6, mt = wv / 6;
  int lo = lane & 15, hi = lane >> 4;
  const f16* ap = gbuf + (size_t)(mt * 16 + lo) * 192 + hi * 16;
  const f16* bp = wout + (nt * 16 + lo) * 192 + hi * 16;
  v8f acc = {};
#pragma unroll
  for (int kc = 0; kc < 6; ++kc)
    acc = wmma16(ldA(ap + kc * 32), *(const v16h*)(bp + kc * 32), acc);
  int n = nt * 16 + lo;
#pragma unroll
  for (int r = 0; r < 8; ++r) {
    int m = mt * 16 + r + 8 * hi;
    out[m * 96 + n] = acc[r] + diff[m * 96 + n];
  }
}

// ---------------------------------------------------------------------------
extern "C" void kernel_launch(void* const* d_in, const int* in_sizes, int n_in,
                              void* d_out, int out_size, void* d_ws, size_t ws_size,
                              hipStream_t stream) {
  const float* x     = (const float*)d_in[0];
  const float* y     = (const float*)d_in[1];
  const float* lnw   = (const float*)d_in[2];
  const float* lnb   = (const float*)d_in[3];
  const float* inpw  = (const float*)d_in[4];
  const float* convw = (const float*)d_in[5];
  const float* convb = (const float*)d_in[6];
  const float* xpw   = (const float*)d_in[7];
  const float* dtw   = (const float*)d_in[8];
  const float* dtb   = (const float*)d_in[9];
  const float* alog  = (const float*)d_in[10];
  const float* Dsp   = (const float*)d_in[11];
  const float* onw   = (const float*)d_in[12];
  const float* onb   = (const float*)d_in[13];
  const float* outpw = (const float*)d_in[14];
  float* out = (float*)d_out;

  char* ws = (char*)d_ws;
  size_t off = 0;
  auto alloc = [&](size_t bytes) -> void* {
    void* p = ws + off;
    off += (bytes + 255) & ~(size_t)255;
    return p;
  };
  float* diff  = (float*)alloc((size_t)8192 * 96 * 4);
  f16*   aln   = (f16*)  alloc((size_t)8192 * 96 * 2);
  f16*   win   = (f16*)  alloc((size_t)36864 * 2);
  f16*   wout  = (f16*)  alloc((size_t)18432 * 2);
  f16*   wxp   = (f16*)  alloc((size_t)36864 * 2);
  f16*   wdt   = (f16*)  alloc((size_t)24576 * 2);
  float* Aneg  = (float*)alloc((size_t)12288 * 4);
  float* xmbuf = (float*)alloc((size_t)2 * 192 * LSEQ * 4);
  float* zbuf  = (float*)alloc((size_t)2 * LSEQ * 192 * 4);
  float* xmem  = (float*)alloc((size_t)2 * LSEQ * 192 * 4);
  f16*   xs0   = (f16*)  alloc((size_t)2 * LSEQ * 192 * 2);
  f16*   xs1   = (f16*)  alloc((size_t)2 * LSEQ * 192 * 2);
  float* xdbl  = (float*)alloc((size_t)8 * 38 * LSEQ * 4);
  float* delta = (float*)alloc((size_t)8 * LSEQ * 192 * 4);
  float* Sb    = (float*)alloc((size_t)8 * NC_ * 3072 * 4);
  float* Hl    = (float*)alloc((size_t)8 * NC_ * 3072 * 4);
  float* Hin   = (float*)alloc((size_t)8 * NC_ * 3072 * 4);
  float* ysm   = (float*)alloc((size_t)8 * LSEQ * 192 * 4);
  f16*   gbuf  = (f16*)  alloc((size_t)8192 * 192 * 2);
  (void)in_sizes; (void)n_in; (void)out_size; (void)ws_size;

  k_pack   <<<504, 256, 0, stream>>>(inpw, outpw, xpw, dtw, alog, win, wout, wxp, wdt, Aneg);
  k_lndiff <<<1024, 256, 0, stream>>>(x, y, lnw, lnb, diff, aln);
  k_inproj <<<1536, 256, 0, stream>>>(aln, win, xmbuf, zbuf);
  k_conv   <<<6144, 256, 0, stream>>>(xmbuf, convw, convb, xmem, xs0, xs1);
  k_xdbl   <<<768, 256, 0, stream>>>(xs0, xs1, wxp, xdbl);
  k_delta  <<<3072, 256, 0, stream>>>(xdbl, wdt, dtb, delta);
  k_scan1  <<<512, 192, 0, stream>>>(xdbl, delta, xmem, Aneg, Sb, Hl);
  k_scan2  <<<96, 256, 0, stream>>>(Sb, Hl, Hin);
  k_scan3  <<<512, 192, 0, stream>>>(xdbl, delta, xmem, Aneg, Hin, Dsp, ysm);
  k_combine<<<1024, 256, 0, stream>>>(ysm, zbuf, onw, onb, gbuf);
  k_outproj<<<384, 256, 0, stream>>>(gbuf, wout, diff, out);
}
